// MultiHeadAttention_2010044695227
// MI455X (gfx1250) — matrix operand order repaired and checked by reading, not compiled, not run on hardware
//
#include <hip/hip_runtime.h>
#include <cstdint>

// ---------------------------------------------------------------------------
// MHA forward for MI455X (gfx1250): bf16 WMMA, DPP softmax reductions,
// double-buffered async global->LDS staging of K/V, exp2-domain softmax,
// packed bf16 conversion.
// D_MODEL=1024, H=16, DEPTH=64, B=2, S=2048.
// ---------------------------------------------------------------------------

typedef __bf16 v16bf __attribute__((ext_vector_type(16)));
typedef __bf16 v8bf  __attribute__((ext_vector_type(8)));
typedef __bf16 v2bf  __attribute__((ext_vector_type(2)));
typedef float  v8f   __attribute__((ext_vector_type(8)));
typedef unsigned v4u __attribute__((ext_vector_type(4)));

#define DM    1024
#define SLEN  2048
#define NH    16
#define DEPTH 64

// log2(e); softmax runs in the exp2 domain (v_exp_f32 is natively base-2).
#define LOG2E     1.4426950408889634f
#define SCALE2    (0.125f * LOG2E)
#define MASKMUL2  (-1.0e9f * LOG2E)

#if defined(__gfx1250__) && __has_builtin(__builtin_amdgcn_global_load_async_to_lds_b128)
#define USE_ASYNC 1
#else
#define USE_ASYNC 0
#endif

#if USE_ASYNC
// The builtin's params are pointers to a 16-byte int vector (per hipcc
// diagnostic): global (AS1) source, LDS (AS3) destination.
typedef int v4i_vs __attribute__((vector_size(16)));
typedef __attribute__((address_space(1))) v4i_vs* async_gptr;
typedef __attribute__((address_space(3))) v4i_vs* async_lptr;
#endif

#if defined(__gfx1250__) && __has_builtin(__builtin_amdgcn_cvt_pk_bf16_f32)
#define HAVE_PK_BF16 1
#else
#define HAVE_PK_BF16 0
#endif

__device__ __forceinline__ float fast_exp2(float x) {
#if defined(__gfx1250__) && __has_builtin(__builtin_amdgcn_exp2f)
  return __builtin_amdgcn_exp2f(x);  // v_exp_f32
#else
  return exp2f(x);
#endif
}

__device__ __forceinline__ __bf16 f2bf(float f) {
  unsigned u = __builtin_bit_cast(unsigned, f);
  unsigned short h = (unsigned short)((u + 0x7FFFu + ((u >> 16) & 1u)) >> 16);
  return __builtin_bit_cast(__bf16, h);
}

// Packed f32x2 -> bf16x2 (one v_cvt_pk_bf16_f32 when available).
__device__ __forceinline__ v2bf pk_bf16(float a, float b) {
#if HAVE_PK_BF16
  auto t = __builtin_amdgcn_cvt_pk_bf16_f32(a, b);
  return __builtin_bit_cast(v2bf, t);
#else
  v2bf r;
  r[0] = f2bf(a);
  r[1] = f2bf(b);
  return r;
#endif
}

// A-matrix fragment (16-bit, 16x32): elements 0..7 <- K=16*half..+7,
// elements 8..15 <- K=16*half+8..+15 (contiguous k, same as B). Two 16B loads, no movs.
__device__ __forceinline__ v16bf frag_a(const __bf16* lo, const __bf16* hi) {
  v8bf l = *(const v8bf*)lo;
  v8bf h = *(const v8bf*)hi;
  return __builtin_shufflevector(l, h, 0, 1, 2, 3, 4, 5, 6, 7,
                                 8, 9, 10, 11, 12, 13, 14, 15);
}

__device__ __forceinline__ v8f wmma_bf16(v16bf a, v16bf b, v8f c) {
  return __builtin_amdgcn_wmma_f32_16x16x32_bf16(false, a, false, b,
                                                 (short)0, c, false, false);
}

// --- 16-lane butterfly reductions in pure VALU via DPP16 -------------------
template <int CTRL>
__device__ __forceinline__ float dppmov(float v) {
  return __builtin_bit_cast(
      float, __builtin_amdgcn_update_dpp(0, __builtin_bit_cast(int, v), CTRL,
                                         0xF, 0xF, true));
}
__device__ __forceinline__ float redmax16(float v) {
  v = fmaxf(v, dppmov<0xB1>(v));   // quad_perm [1,0,3,2]  (^1)
  v = fmaxf(v, dppmov<0x4E>(v));   // quad_perm [2,3,0,1]  (^2)
  v = fmaxf(v, dppmov<0x141>(v));  // row_half_mirror      (^4 eff)
  v = fmaxf(v, dppmov<0x140>(v));  // row_mirror           (^8 eff)
  return v;
}
__device__ __forceinline__ float redsum16(float v) {
  v += dppmov<0xB1>(v);
  v += dppmov<0x4E>(v);
  v += dppmov<0x141>(v);
  v += dppmov<0x140>(v);
  return v;
}

// --- async global -> LDS (CDNA5 ASYNCcnt path) -----------------------------
__device__ __forceinline__ void async_b128(const __bf16* g, __bf16* l) {
#if USE_ASYNC
  // Low 32 bits of a generic LDS pointer are the LDS byte address (ISA 10.2).
  __builtin_amdgcn_global_load_async_to_lds_b128(
      (async_gptr)(uintptr_t)g, (async_lptr)(unsigned)(uintptr_t)l, 0, 0);
#else
  *(v8bf*)l = *(const v8bf*)g;
  *(v8bf*)(l + 8) = *(const v8bf*)(g + 8);
#endif
}
__device__ __forceinline__ void wait_async_4() {
#if USE_ASYNC
#if __has_builtin(__builtin_amdgcn_s_wait_asynccnt)
  __builtin_amdgcn_s_wait_asynccnt(4);
#else
  asm volatile("s_wait_asynccnt 0x4" ::: "memory");
#endif
#endif
}
__device__ __forceinline__ void wait_async_0() {
#if USE_ASYNC
#if __has_builtin(__builtin_amdgcn_s_wait_asynccnt)
  __builtin_amdgcn_s_wait_asynccnt(0);
#else
  asm volatile("s_wait_asynccnt 0x0" ::: "memory");
#endif
#endif
}

// ---------------------------------------------------------------------------
// GEMM: C[4096x1024] = X[4096x1024] @ W[1024x1024] + bias
// Block: 256 threads = 8 waves, tile 64(M) x 32(N), K-chunks of 32.
// mode 0: bf16 out, head-split [B,H,S,64]
// mode 2: bf16 out, head-split transposed [B,H,64,S]   (for V)
// mode 3: f32 out, row-major [4096,1024]               (final projection)
// ---------------------------------------------------------------------------
template <bool XF32>
__global__ __launch_bounds__(256) void gemm_bias_kernel(
    const void* __restrict__ Xv, const float* __restrict__ W,
    const float* __restrict__ bias, void* __restrict__ out, int mode) {
  __shared__ __align__(32) __bf16 Xs[64][32];   // [m][k]
  __shared__ __align__(32) __bf16 WsT[32][32];  // [n][k]

  const int tid  = threadIdx.x;
  const int lane = tid & 31, wave = tid >> 5;
  const int wm = wave & 3, wn = wave >> 2;
  const int nl = lane & 15, half = lane >> 4;
  const int m0 = blockIdx.x * 64, n0 = blockIdx.y * 32;

  v8f c;
#pragma unroll
  for (int r = 0; r < 8; ++r) c[r] = 0.f;

  for (int k0 = 0; k0 < DM; k0 += 32) {
    {  // X tile: 2048 bf16 elems, 8 per thread (contiguous in k)
      const int i0 = tid * 8;
      const int m = i0 >> 5, kk = i0 & 31;
      if (XF32) {
        const float* src = (const float*)Xv + (size_t)(m0 + m) * DM + k0 + kk;
        float4 p0 = *(const float4*)src;
        float4 p1 = *(const float4*)(src + 4);
        v4u pack;  // 4x v_cvt_pk_bf16_f32 + one 16B LDS store
        pack[0] = __builtin_bit_cast(unsigned, pk_bf16(p0.x, p0.y));
        pack[1] = __builtin_bit_cast(unsigned, pk_bf16(p0.z, p0.w));
        pack[2] = __builtin_bit_cast(unsigned, pk_bf16(p1.x, p1.y));
        pack[3] = __builtin_bit_cast(unsigned, pk_bf16(p1.z, p1.w));
        *(v4u*)&Xs[m][kk] = pack;
      } else {
        const __bf16* src = (const __bf16*)Xv + (size_t)(m0 + m) * DM + k0 + kk;
        *(v8bf*)&Xs[m][kk] = *(const v8bf*)src;
      }
    }
    {  // W tile 32(k) x 32(n), stored transposed -> WsT[n][k]
      const int i0 = tid * 4;
      const int kr = i0 >> 5, n = i0 & 31;
      float4 w4 = *(const float4*)(W + (size_t)(k0 + kr) * DM + n0 + n);
      v2bf a01 = pk_bf16(w4.x, w4.y);
      v2bf a23 = pk_bf16(w4.z, w4.w);
      WsT[n + 0][kr] = a01[0];
      WsT[n + 1][kr] = a01[1];
      WsT[n + 2][kr] = a23[0];
      WsT[n + 3][kr] = a23[1];
    }
    __syncthreads();
    v16bf a = frag_a(&Xs[wm * 16 + nl][half * 16],
                     &Xs[wm * 16 + nl][half * 16 + 8]);
    v16bf b = *(const v16bf*)&WsT[wn * 16 + nl][half * 16];
    c = wmma_bf16(a, b, c);
    __syncthreads();
  }

  const int n_g = n0 + wn * 16 + nl;
  const float badd = bias[n_g];
#pragma unroll
  for (int r = 0; r < 8; ++r) {
    const int m_g = m0 + wm * 16 + r + 8 * half;  // C-layout row
    const float val = c[r] + badd;
    if (mode == 3) {
      ((float*)out)[(size_t)m_g * DM + n_g] = val;
    } else {
      const int bb = m_g >> 11, s = m_g & (SLEN - 1);
      const int h = n_g >> 6, d = n_g & (DEPTH - 1);
      __bf16* o = (__bf16*)out;
      size_t idx = (mode == 0)
          ? ((((size_t)(bb * NH + h)) * SLEN + s) * DEPTH + d)
          : ((((size_t)(bb * NH + h)) * DEPTH + d) * SLEN + s);
      o[idx] = f2bf(val);
    }
  }
}

// ---------------------------------------------------------------------------
// Flash attention. Block = 8 waves, each owns a 16-row Q tile (128 q rows).
// kv streamed in 64-wide chunks; K chunk (64x64) and V chunk (64x64, already
// d-major) staged in LDS via async copies, double-buffered, shared by all
// 8 waves. 16 WMMAs per wave per chunk. Softmax in exp2 domain.
// ---------------------------------------------------------------------------
__global__ __launch_bounds__(256) void attn_kernel(
    const void* __restrict__ Qv, const void* __restrict__ Kv,
    const void* __restrict__ Vv, const float* __restrict__ maskp,
    void* __restrict__ AOv) {
  __shared__ __align__(32) __bf16 Ks[2][64][64];  // [buf][kv_local][d]  16KB
  __shared__ __align__(32) __bf16 Vs[2][64][64];  // [buf][d][kv_local]  16KB
  __shared__ __align__(32) __bf16 Ps[8][16][64];  // per-wave P tiles    16KB

  const int tid  = threadIdx.x;
  const int lane = tid & 31, wave = tid >> 5;
  const int nl = lane & 15, half = lane >> 4;
  const int bh = blockIdx.y, b = bh >> 4, h = bh & (NH - 1);
  const int q0 = blockIdx.x * 128 + wave * 16;

  const __bf16* Q = (const __bf16*)Qv + (size_t)bh * SLEN * DEPTH;
  const __bf16* K = (const __bf16*)Kv + (size_t)bh * SLEN * DEPTH;
  const __bf16* V = (const __bf16*)Vv + (size_t)bh * DEPTH * SLEN;
  const float*  mk = maskp + (size_t)b * SLEN;
  __bf16* AO = (__bf16*)AOv;

  // Q fragments (A-matrix): 16 rows x 64 depth = two K=32 fragments.
  const __bf16* qp = Q + (size_t)(q0 + nl) * DEPTH;
  v16bf qf0 = frag_a(qp + half * 16,      qp + 8 + half * 16);
  v16bf qf1 = frag_a(qp + 32 + half * 16, qp + 40 + half * 16);

  // Per-thread staging addresses for one 64-kv chunk (4 x b128 per thread):
  // K chunk: contiguous 8KB; thread t covers bf16 elems [t*16, t*16+16).
  // V chunk: 64 d-rows x 128B; 4 threads per row, 32B each.
  const int vrow = tid >> 2, vpart = tid & 3;

  float mi[8], li[8];
  v8f acc[4];
#pragma unroll
  for (int r = 0; r < 8; ++r) { mi[r] = -3.0e38f; li[r] = 0.f; }
#pragma unroll
  for (int i = 0; i < 4; ++i)
#pragma unroll
    for (int r = 0; r < 8; ++r) acc[i][r] = 0.f;

  // Prologue: stage chunk 0 into buffer 0.
  {
    const __bf16* kg = K + tid * 16;
    async_b128(kg, &Ks[0][0][0] + tid * 16);
    async_b128(kg + 8, &Ks[0][0][0] + tid * 16 + 8);
    const __bf16* vg = V + (size_t)vrow * SLEN + vpart * 16;
    async_b128(vg, &Vs[0][vrow][vpart * 16]);
    async_b128(vg + 8, &Vs[0][vrow][vpart * 16] + 8);
  }

  for (int kv = 0; kv < SLEN; kv += 64) {
    const int buf = (kv >> 6) & 1;
    if (kv + 64 < SLEN) {  // prefetch next chunk into the other buffer
      const __bf16* kg = K + (size_t)(kv + 64) * DEPTH + tid * 16;
      async_b128(kg, &Ks[buf ^ 1][0][0] + tid * 16);
      async_b128(kg + 8, &Ks[buf ^ 1][0][0] + tid * 16 + 8);
      const __bf16* vg = V + (size_t)vrow * SLEN + (kv + 64) + vpart * 16;
      async_b128(vg, &Vs[buf ^ 1][vrow][vpart * 16]);
      async_b128(vg + 8, &Vs[buf ^ 1][vrow][vpart * 16] + 8);
      wait_async_4();  // the 4 just-issued may remain outstanding
    } else {
      wait_async_0();
    }
    __syncthreads();

    // --- logits (exp2 domain): 4 score tiles x (K-depth 64 = 2 WMMAs)
    v8f st[4];
    float madd[4];
#pragma unroll
    for (int t = 0; t < 4; ++t) {
      const __bf16* kr = &Ks[buf][t * 16 + nl][0];
#pragma unroll
      for (int r = 0; r < 8; ++r) st[t][r] = 0.f;
      st[t] = wmma_bf16(qf0, *(const v16bf*)(kr + half * 16), st[t]);
      st[t] = wmma_bf16(qf1, *(const v16bf*)(kr + 32 + half * 16), st[t]);
      madd[t] = mk[kv + t * 16 + nl] * MASKMUL2;
#pragma unroll
      for (int r = 0; r < 8; ++r) st[t][r] = st[t][r] * SCALE2 + madd[t];
    }

    // --- online softmax (rows r+8*half; DPP butterflies stay in-half)
#pragma unroll
    for (int r = 0; r < 8; ++r) {
      float mx = fmaxf(fmaxf(st[0][r], st[1][r]), fmaxf(st[2][r], st[3][r]));
      mx = redmax16(mx);
      float mnew = fmaxf(mi[r], mx);
      float alpha = fast_exp2(mi[r] - mnew);
      float p0 = fast_exp2(st[0][r] - mnew);
      float p1 = fast_exp2(st[1][r] - mnew);
      float p2 = fast_exp2(st[2][r] - mnew);
      float p3 = fast_exp2(st[3][r] - mnew);
      li[r] = li[r] * alpha + redsum16((p0 + p1) + (p2 + p3));
      mi[r] = mnew;
#pragma unroll
      for (int i = 0; i < 4; ++i) acc[i][r] *= alpha;
      const int pr = r + 8 * half;  // lane-transpose P via per-wave LDS tile
      v2bf p01 = pk_bf16(p0, p1);
      v2bf p23 = pk_bf16(p2, p3);
      Ps[wave][pr][nl]      = p01[0];
      Ps[wave][pr][16 + nl] = p01[1];
      Ps[wave][pr][32 + nl] = p23[0];
      Ps[wave][pr][48 + nl] = p23[1];
    }

    // --- O += P(16x64) @ V(64x64): 2 K-fragments x 4 d-tiles = 8 WMMAs
#pragma unroll
    for (int kt = 0; kt < 2; ++kt) {
      v16bf pf = frag_a(&Ps[wave][nl][kt * 32 + half * 16],
                        &Ps[wave][nl][kt * 32 + half * 16 + 8]);
#pragma unroll
      for (int nt = 0; nt < 4; ++nt) {
        const __bf16* vr = &Vs[buf][nt * 16 + nl][kt * 32 + half * 16];
        acc[nt] = wmma_bf16(pf, *(const v16bf*)vr, acc[nt]);
      }
    }
    __syncthreads();  // everyone done with buf before it is restaged
  }

#pragma unroll
  for (int r = 0; r < 8; ++r) {
    const float inv = 1.0f / li[r];
    const int row_g = b * SLEN + q0 + r + 8 * half;  // [B*S] row
    v2bf o01 = pk_bf16(acc[0][r] * inv, acc[1][r] * inv);
    v2bf o23 = pk_bf16(acc[2][r] * inv, acc[3][r] * inv);
    __bf16* dst = AO + (size_t)row_g * DM + h * DEPTH + nl;
    dst[0]  = o01[0];
    dst[16] = o01[1];
    dst[32] = o23[0];
    dst[48] = o23[1];
  }
}

// ---------------------------------------------------------------------------
// Launch: 3 projections -> attention -> output projection.
// inputs: 0=v 1=k 2=q 3=mask 4=wq 5=bq 6=wk 7=bk 8=wv 9=bv 10=wo 11=bo
// ws: Qh(8MB) | Kh(8MB) | Vt(8MB) | attn_out(8MB)
// ---------------------------------------------------------------------------
extern "C" void kernel_launch(void* const* d_in, const int* in_sizes, int n_in,
                              void* d_out, int out_size, void* d_ws,
                              size_t ws_size, hipStream_t stream) {
  (void)in_sizes; (void)n_in; (void)out_size; (void)ws_size;
  const float* vp = (const float*)d_in[0];
  const float* kp = (const float*)d_in[1];
  const float* qp = (const float*)d_in[2];
  const float* mask = (const float*)d_in[3];
  const float* wq = (const float*)d_in[4];
  const float* bq = (const float*)d_in[5];
  const float* wk = (const float*)d_in[6];
  const float* bk = (const float*)d_in[7];
  const float* wv = (const float*)d_in[8];
  const float* bv = (const float*)d_in[9];
  const float* wo = (const float*)d_in[10];
  const float* bo = (const float*)d_in[11];

  const size_t HB = (size_t)4096 * 1024 * 2;  // bytes per bf16 [4096x1024]
  char* ws = (char*)d_ws;
  void* Qh = ws;
  void* Kh = ws + HB;
  void* Vt = ws + 2 * HB;
  void* AO = ws + 3 * HB;

  dim3 gg(64, 32), gb(256);
  gemm_bias_kernel<true><<<gg, gb, 0, stream>>>(qp, wq, bq, Qh, 0);
  gemm_bias_kernel<true><<<gg, gb, 0, stream>>>(kp, wk, bk, Kh, 0);
  gemm_bias_kernel<true><<<gg, gb, 0, stream>>>(vp, wv, bv, Vt, 2);
  attn_kernel<<<dim3(16, 32), 256, 0, stream>>>(Qh, Kh, Vt, mask, AO);
  gemm_bias_kernel<false><<<gg, gb, 0, stream>>>(AO, wo, bo, d_out, 3);
}
